// RNNCell_43911745634884
// MI455X (gfx1250) — matrix operand order repaired and checked by reading, not compiled, not run on hardware
//
#include <hip/hip_runtime.h>

// ---------------------------------------------------------------------------
// Multi-layer leaky-ReLU RNN, B=64 T=256 I=512 H=1024 L=6.
// Layers independent (each reads raw x[t]); batch rows independent.
// One workgroup per (layer, batch-half): h state double-buffered in LDS,
// per-step GEMM [32 x 1536] x [1536 x 1024] via v_wmma_f32_16x16x32_bf16.
// 512 threads = 16 waves; wave w owns col-tiles 4w..4w+3, both row-tiles.
// A zero offset is laundered through asm per timestep so the compiler cannot
// hoist the (t-invariant) weight loads out of the scan loop (which spilled to
// scratch), while keeping pointer provenance so loads stay GLOBAL not FLAT.
// ---------------------------------------------------------------------------

typedef __attribute__((ext_vector_type(16))) __bf16 v16bf;
typedef __attribute__((ext_vector_type(8)))  float  v8f;

#define B_  64
#define T_  256
#define I_  512
#define H_  1024
#define L_  6
#define NEG_SLOPE 0.01f
#define PITCH (H_ + 8)                 // bf16 elems per LDS row (bank spread)
#define SMEM_BYTES (2 * 32 * PITCH * 2)

union FragU { v16bf v; uint4 q[2]; };

// ---- pre-pass: fp32 -> bf16 conversions -----------------------------------

__global__ void cvt_x_kernel(const float* __restrict__ x, __bf16* __restrict__ xb) {
    size_t idx = (size_t)blockIdx.x * blockDim.x + threadIdx.x;
    if (idx >= (size_t)B_ * T_ * I_) return;
    int i = (int)(idx % I_);
    int t = (int)((idx / I_) % T_);
    int b = (int)(idx / ((size_t)I_ * T_));
    // [B,T,I] -> [T,B,I]
    xb[((size_t)t * B_ + b) * I_ + i] = (__bf16)x[idx];
}

__global__ void cvt_w_kernel(const float* __restrict__ src, __bf16* __restrict__ dst, int n) {
    int idx = blockIdx.x * blockDim.x + threadIdx.x;
    if (idx < n) dst[idx] = (__bf16)src[idx];
}

__global__ void bias_kernel(const float* __restrict__ bih, const float* __restrict__ bhh,
                            float* __restrict__ bc, int n) {
    int idx = blockIdx.x * blockDim.x + threadIdx.x;
    if (idx < n) bc[idx] = bih[idx] + bhh[idx];
}

// ---- recurrent kernel ------------------------------------------------------

__global__ __launch_bounds__(512, 1)
void rnn_scan_kernel(const __bf16* __restrict__ xb,    // [T][B][I]
                     const __bf16* __restrict__ wih,   // [L][H][I]  ([N][K])
                     const __bf16* __restrict__ whh,   // [L][H][H]  ([N][K])
                     const float*  __restrict__ biasc, // [L][H]
                     const float*  __restrict__ h0,    // [L][B][H] fp32
                     float* __restrict__ out,          // [B][T][H]
                     float* __restrict__ hT)           // [L][B][H]
{
    extern __shared__ __bf16 smem[];                   // [2][32][PITCH]
    __bf16* hbuf0 = smem;
    __bf16* hbuf1 = smem + 32 * PITCH;

    const int layer = blockIdx.x >> 1;
    const int bBase = (blockIdx.x & 1) * 32;           // batch-half
    const int tid   = threadIdx.x;
    const int lane  = tid & 31;                        // wave32
    const int wave  = tid >> 5;                        // 0..15
    const int lrow  = lane & 15;                       // m (A) / n (B,C) in tile
    const int lhi   = lane >> 4;                       // half-wave select

    // load h_0 -> LDS buffer 0 (bf16)
    const float* h0l = h0 + ((size_t)layer * B_ + bBase) * H_;
    for (int e = tid; e < 32 * H_; e += 512) {
        int r = e >> 10, c = e & (H_ - 1);
        hbuf0[r * PITCH + c] = (__bf16)h0l[e];
    }
    __syncthreads();

    const __bf16* wihL = wih + (size_t)layer * H_ * I_;
    const __bf16* whhL = whh + (size_t)layer * H_ * H_;

    // this wave owns column tiles 4*wave .. 4*wave+3
    float biasv[4];
#pragma unroll
    for (int nj = 0; nj < 4; ++nj)
        biasv[nj] = biasc[layer * H_ + (wave * 4 + nj) * 16 + lrow];

    for (int t = 0; t < T_; ++t) {
        __bf16* cur = (t & 1) ? hbuf1 : hbuf0;
        __bf16* nxt = (t & 1) ? hbuf0 : hbuf1;

        // Launder a zero element-offset each timestep: blocks loop-invariant
        // hoisting of the weight loads across t (which caused scratch spills)
        // while preserving the global address space (global_load, not flat).
        size_t woff = 0;
        asm volatile("" : "+s"(woff));
        const __bf16* wihT = wihL + woff;
        const __bf16* whhT = whhL + woff;

        v8f acc[2][4];
#pragma unroll
        for (int mi = 0; mi < 2; ++mi)
#pragma unroll
            for (int nj = 0; nj < 4; ++nj)
#pragma unroll
                for (int e = 0; e < 8; ++e) acc[mi][nj][e] = biasv[nj];

        // ---- K part 1: x[t] @ W_ih^T, K = 512 (16 k-steps of 32) ----------
        const __bf16* xrow = xb + ((size_t)t * B_ + bBase) * I_;
        for (int kk = 0; kk < I_ / 32; ++kk) {
            const int kb = kk * 32;
            FragU afr[2], bfr[4];
#pragma unroll
            for (int mi = 0; mi < 2; ++mi) {
                const uint4* p = (const uint4*)(xrow + (size_t)(mi * 16 + lrow) * I_ + kb + lhi * 8);
                afr[mi].q[0] = p[0];                   // K = kb + lhi*8 + [0..7]
                afr[mi].q[1] = p[2];                   // K = +16
            }
#pragma unroll
            for (int nj = 0; nj < 4; ++nj) {
                int n = (wave * 4 + nj) * 16 + lrow;
                const uint4* p = (const uint4*)(wihT + (size_t)n * I_ + kb + lhi * 8);
                bfr[nj].q[0] = p[0];                   // K = kb + lhi*8 + [0..7]
                bfr[nj].q[1] = p[2];                   // K = +16
            }
#pragma unroll
            for (int mi = 0; mi < 2; ++mi)
#pragma unroll
                for (int nj = 0; nj < 4; ++nj)
                    acc[mi][nj] = __builtin_amdgcn_wmma_f32_16x16x32_bf16(
                        false, afr[mi].v, false, bfr[nj].v,
                        (short)0, acc[mi][nj], false, false);
        }

        // ---- K part 2: h @ W_hh^T, K = 1024 (32 k-steps, A from LDS) ------
        for (int kk = 0; kk < H_ / 32; ++kk) {
            const int kb = kk * 32;
            FragU afr[2], bfr[4];
#pragma unroll
            for (int mi = 0; mi < 2; ++mi) {
                const uint4* p = (const uint4*)(cur + (size_t)(mi * 16 + lrow) * PITCH + kb + lhi * 8);
                afr[mi].q[0] = p[0];
                afr[mi].q[1] = p[2];
            }
#pragma unroll
            for (int nj = 0; nj < 4; ++nj) {
                int n = (wave * 4 + nj) * 16 + lrow;
                const uint4* p = (const uint4*)(whhT + (size_t)n * H_ + kb + lhi * 8);
                bfr[nj].q[0] = p[0];
                bfr[nj].q[1] = p[2];
            }
#pragma unroll
            for (int mi = 0; mi < 2; ++mi)
#pragma unroll
                for (int nj = 0; nj < 4; ++nj)
                    acc[mi][nj] = __builtin_amdgcn_wmma_f32_16x16x32_bf16(
                        false, afr[mi].v, false, bfr[nj].v,
                        (short)0, acc[mi][nj], false, false);
        }

        // ---- leaky-ReLU + writeback (LDS next buffer, global outputs) -----
#pragma unroll
        for (int mi = 0; mi < 2; ++mi) {
#pragma unroll
            for (int nj = 0; nj < 4; ++nj) {
                const int n = (wave * 4 + nj) * 16 + lrow;
#pragma unroll
                for (int e = 0; e < 8; ++e) {
                    float v = acc[mi][nj][e];
                    v = (v > 0.0f) ? v : v * NEG_SLOPE;
                    const int rloc = mi * 16 + lhi * 8 + e;     // D layout rows
                    nxt[rloc * PITCH + n] = (__bf16)v;
                    if (layer == L_ - 1)
                        out[((size_t)(bBase + rloc)) * T_ * H_ + (size_t)t * H_ + n] = v;
                    if (t == T_ - 1)
                        hT[((size_t)layer * B_ + bBase + rloc) * H_ + n] = v;
                }
            }
        }
        __syncthreads();   // writes to nxt done before next step reads it
    }
}

// ---------------------------------------------------------------------------

extern "C" void kernel_launch(void* const* d_in, const int* in_sizes, int n_in,
                              void* d_out, int out_size, void* d_ws, size_t ws_size,
                              hipStream_t stream) {
    const float* x   = (const float*)d_in[0];   // [B,T,I]
    const float* h0  = (const float*)d_in[1];   // [L,B,H]
    const float* wih = (const float*)d_in[2];   // [L,H,I]
    const float* whh = (const float*)d_in[3];   // [L,H,H]
    const float* bih = (const float*)d_in[4];   // [L,H]
    const float* bhh = (const float*)d_in[5];   // [L,H]

    float* out = (float*)d_out;                          // [B,T,H]
    float* hT  = out + (size_t)B_ * T_ * H_;             // [L,B,H]

    // workspace carve-up (bytes)
    char* ws = (char*)d_ws;
    __bf16* xb     = (__bf16*)(ws);                                       // 16 MB
    __bf16* wih_bf = (__bf16*)(ws + (size_t)T_ * B_ * I_ * 2);            //  6 MB
    __bf16* whh_bf = (__bf16*)((char*)wih_bf + (size_t)L_ * H_ * I_ * 2); // 12 MB
    float*  biasc  = (float*)((char*)whh_bf + (size_t)L_ * H_ * H_ * 2);  // 24 KB

    const int nx   = B_ * T_ * I_;
    const int nwih = L_ * H_ * I_;
    const int nwhh = L_ * H_ * H_;
    const int nb   = L_ * H_;

    cvt_x_kernel<<<(nx + 255) / 256, 256, 0, stream>>>(x, xb);
    cvt_w_kernel<<<(nwih + 255) / 256, 256, 0, stream>>>(wih, wih_bf, nwih);
    cvt_w_kernel<<<(nwhh + 255) / 256, 256, 0, stream>>>(whh, whh_bf, nwhh);
    bias_kernel<<<(nb + 255) / 256, 256, 0, stream>>>(bih, bhh, biasc, nb);

    hipFuncSetAttribute((const void*)rnn_scan_kernel,
                        hipFuncAttributeMaxDynamicSharedMemorySize, SMEM_BYTES);

    rnn_scan_kernel<<<L_ * 2, 512, SMEM_BYTES, stream>>>(
        xb, wih_bf, whh_bf, biasc, h0, out, hT);
}
